// TemporalFGNNIwithAutoformer_75050258530308
// MI455X (gfx1250) — compile-verified, matrix operand order checked by reading, not run on hardware
//
#include <hip/hip_runtime.h>
#include <hip/hip_bf16.h>
#include <math.h>

typedef __attribute__((ext_vector_type(16))) _Float16 v16h;
typedef __attribute__((ext_vector_type(8)))  _Float16 v8h;
typedef __attribute__((ext_vector_type(8)))  float    v8f;

#define TPB 256

__device__ __forceinline__ float gelu_exact(float x) {
    return 0.5f * x * (1.0f + erff(x * 0.70710678118654752f));
}

// ---------------------------------------------------------------------------
// WMMA GEMM:  C[z][m][n] = act( A[z][m][:] . B[z][n][:] + bias[n] + Res[z][m][n] )
// A: (batch, M, K) f32 row-major (stride bsA), B: (batch, N, K) f32 (stride bsB,
// 0 for shared weights). Block tile 128(M) x 128(N), K step 32, 8 waves/block,
// each wave computes 64x32 via 8x v_wmma_f32_16x16x32_f16 per K chunk
// (4 A-fragments reused across 2 B-fragments).
// ---------------------------------------------------------------------------
__global__ __launch_bounds__(TPB) void k_gemm_nt(
    const float* __restrict__ A, long long bsA,
    const float* __restrict__ Bm, long long bsB,
    const float* __restrict__ bias,
    const float* __restrict__ Res, long long bsR,
    float* __restrict__ C, long long bsC,
    int M, int N, int K, int act)
{
    __shared__ _Float16 sA[128 * 40];
    __shared__ _Float16 sB[128 * 40];

    const int z   = blockIdx.z;
    const float* Az = A + (size_t)z * bsA;
    const float* Bz = Bm + (size_t)z * bsB;
    const int bm = blockIdx.y * 128;
    const int bn = blockIdx.x * 128;

    const int tid   = threadIdx.x;
    const int wave  = tid >> 5;
    const int lane  = tid & 31;
    const int wm    = wave >> 2;      // 0..1 -> M offset wm*64
    const int wn    = wave & 3;       // 0..3 -> N offset wn*32
    const int lrow  = lane & 15;
    const int lhalf = lane >> 4;
    const int k0    = lhalf * 8;

    // staging mapping (same for A and B): 128 rows x 32 cols, 2 threads/row
    const int srow  = tid >> 1;
    const int scol0 = (tid & 1) * 16;

    v8f acc[4][2];
    #pragma unroll
    for (int i = 0; i < 4; i++)
        #pragma unroll
        for (int j = 0; j < 2; j++)
            acc[i][j] = v8f{0.f,0.f,0.f,0.f,0.f,0.f,0.f,0.f};

    for (int kb = 0; kb < K; kb += 32) {
        // --- stage A tile ---
        {
            const int gr = bm + srow;
            float vals[16];
            if (gr < M) {
                const float4* p = (const float4*)(Az + (size_t)gr * K + kb + scol0);
                #pragma unroll
                for (int i = 0; i < 4; i++) {
                    float4 f = p[i];
                    vals[4*i+0] = f.x; vals[4*i+1] = f.y; vals[4*i+2] = f.z; vals[4*i+3] = f.w;
                }
                if (kb + 32 < K)  // prefetch next K chunk -> global_prefetch_b8
                    __builtin_prefetch(Az + (size_t)gr * K + kb + 32 + scol0, 0, 1);
            } else {
                #pragma unroll
                for (int i = 0; i < 16; i++) vals[i] = 0.f;
            }
            #pragma unroll
            for (int i = 0; i < 16; i++) sA[srow * 40 + scol0 + i] = (_Float16)vals[i];
        }
        // --- stage B tile ---
        {
            const int gn = bn + srow;
            float vals[16];
            if (gn < N) {
                const float4* p = (const float4*)(Bz + (size_t)gn * K + kb + scol0);
                #pragma unroll
                for (int i = 0; i < 4; i++) {
                    float4 f = p[i];
                    vals[4*i+0] = f.x; vals[4*i+1] = f.y; vals[4*i+2] = f.z; vals[4*i+3] = f.w;
                }
                if (kb + 32 < K)
                    __builtin_prefetch(Bz + (size_t)gn * K + kb + 32 + scol0, 0, 1);
            } else {
                #pragma unroll
                for (int i = 0; i < 16; i++) vals[i] = 0.f;
            }
            #pragma unroll
            for (int i = 0; i < 16; i++) sB[srow * 40 + scol0 + i] = (_Float16)vals[i];
        }
        __syncthreads();

        // B fragments (32x16 K x N each): lane col = lrow, K halves {k0..+7, k0+16..+23}
        v16h bf[2];
        #pragma unroll
        for (int ns = 0; ns < 2; ns++) {
            v8h bl = *(const v8h*)&sB[(wn * 32 + ns * 16 + lrow) * 40 + k0];
            v8h bh = *(const v8h*)&sB[(wn * 32 + ns * 16 + lrow) * 40 + k0 + 16];
            bf[ns] = __builtin_shufflevector(bl, bh, 0,1,2,3,4,5,6,7,8,9,10,11,12,13,14,15);
        }
        #pragma unroll
        for (int i = 0; i < 4; i++) {
            const int r = wm * 64 + i * 16 + lrow;
            v8h al = *(const v8h*)&sA[r * 40 + k0];
            v8h ah = *(const v8h*)&sA[r * 40 + k0 + 16];
            v16h af = __builtin_shufflevector(al, ah, 0,1,2,3,4,5,6,7,8,9,10,11,12,13,14,15);
            acc[i][0] = __builtin_amdgcn_wmma_f32_16x16x32_f16(
                false, af, false, bf[0], (short)0, acc[i][0], false, false);
            acc[i][1] = __builtin_amdgcn_wmma_f32_16x16x32_f16(
                false, af, false, bf[1], (short)0, acc[i][1], false, false);
        }
        __syncthreads();
    }

    // epilogue: C layout lane col = lrow; VGPR j -> row 8*lhalf + j within subtile
    #pragma unroll
    for (int ns = 0; ns < 2; ns++) {
        const int c = bn + wn * 32 + ns * 16 + lrow;
        if (c < N) {
            const float bv = bias ? bias[c] : 0.f;
            #pragma unroll
            for (int i = 0; i < 4; i++) {
                const int rbase = bm + wm * 64 + i * 16 + lhalf * 8;
                #pragma unroll
                for (int j = 0; j < 8; j++) {
                    const int r = rbase + j;
                    if (r < M) {
                        float v = acc[i][ns][j] + bv;
                        if (Res) v += Res[(size_t)z * bsR + (size_t)r * N + c];
                        if (act == 1) v = gelu_exact(v);
                        C[(size_t)z * bsC + (size_t)r * N + c] = v;
                    }
                }
            }
        }
    }
}

// ---------------------------------------------------------------------------
// series decomposition: trend = moving mean (k=25, edge clamped), season = x - trend
// ---------------------------------------------------------------------------
__global__ void k_decomp(const float* __restrict__ x, float* __restrict__ season,
                         float* __restrict__ trend, int Bn, int Lx, int C)
{
    long long idx = (long long)blockIdx.x * TPB + threadIdx.x;
    long long total = (long long)Bn * Lx * C;
    if (idx >= total) return;
    int c = (int)(idx % C);
    int l = (int)((idx / C) % Lx);
    int b = (int)(idx / ((long long)C * Lx));
    float s = 0.f;
    #pragma unroll
    for (int j = -12; j <= 12; j++) {
        int t = l + j;
        t = t < 0 ? 0 : (t >= Lx ? Lx - 1 : t);
        s += x[((size_t)b * Lx + t) * C + c];
    }
    float m = s * (1.f / 25.f);
    trend[idx]  = m;
    season[idx] = x[idx] - m;
}

// mean over axis L: (B, Lx, C) -> (B, C)
__global__ void k_colmean(const float* __restrict__ x, float* __restrict__ mu,
                          int Bn, int Lx, int C)
{
    int idx = blockIdx.x * TPB + threadIdx.x;
    if (idx >= Bn * C) return;
    int c = idx % C, b = idx / C;
    float s = 0.f;
    for (int l = 0; l < Lx; l++) s += x[((size_t)b * Lx + l) * C + c];
    mu[idx] = s / (float)Lx;
}

__global__ void k_submean(const float* __restrict__ x, const float* __restrict__ mu,
                          float* __restrict__ o, int Bn, int Lx, int C)
{
    long long idx = (long long)blockIdx.x * TPB + threadIdx.x;
    long long total = (long long)Bn * Lx * C;
    if (idx >= total) return;
    int c = (int)(idx % C);
    int b = (int)(idx / ((long long)C * Lx));
    o[idx] = x[idx] - mu[b * C + c];
}

// trend/seasonal init: concat(label tail, mean / zeros)
__global__ void k_init_td(const float* __restrict__ tinit, const float* __restrict__ sinit,
                          const float* __restrict__ xmean, float* __restrict__ trend,
                          float* __restrict__ seasonal, int Bn, int L, int C, int Ld, int LAB)
{
    long long idx = (long long)blockIdx.x * TPB + threadIdx.x;
    long long total = (long long)Bn * Ld * C;
    if (idx >= total) return;
    int n = (int)(idx % C);
    int t = (int)((idx / C) % Ld);
    int b = (int)(idx / ((long long)C * Ld));
    if (t < LAB) {
        size_t src = ((size_t)b * L + (L - LAB) + t) * C + n;
        trend[idx]    = tinit[src];
        seasonal[idx] = sinit[src];
    } else {
        trend[idx]    = xmean[b * C + n];
        seasonal[idx] = 0.f;
    }
}

// circular conv im2col: dst[b][l][c*3+j] = src[b][(l-1+j) mod Lx][c]
__global__ void k_build_xs(const float* __restrict__ src, float* __restrict__ dst,
                           int Bn, int Lx, int C)
{
    long long idx = (long long)blockIdx.x * TPB + threadIdx.x;
    long long total = (long long)Bn * Lx * C * 3;
    if (idx >= total) return;
    int j = (int)(idx % 3);
    int c = (int)((idx / 3) % C);
    int l = (int)((idx / (3LL * C)) % Lx);
    int b = (int)(idx / (3LL * C * Lx));
    int t = l - 1 + j;
    if (t < 0) t += Lx;
    if (t >= Lx) t -= Lx;
    dst[idx] = src[((size_t)b * Lx + t) * C + c];
}

// LayerNorm over last dim (D=512): one wave per row
__global__ void k_ln(const float* __restrict__ x, const float* __restrict__ g,
                     const float* __restrict__ bb, float* __restrict__ o, int rows, int D)
{
    int row  = blockIdx.x * 8 + (threadIdx.x >> 5);
    int lane = threadIdx.x & 31;
    if (row >= rows) return;
    const float* xr = x + (size_t)row * D;
    float s = 0.f;
    for (int i = lane; i < D; i += 32) s += xr[i];
    for (int off = 16; off > 0; off >>= 1) s += __shfl_xor(s, off, 32);
    float mu = s / (float)D;
    float v = 0.f;
    for (int i = lane; i < D; i += 32) { float d = xr[i] - mu; v += d * d; }
    for (int off = 16; off > 0; off >>= 1) v += __shfl_xor(v, off, 32);
    float inv = rsqrtf(v / (float)D + 1e-5f);
    float* orow = o + (size_t)row * D;
    for (int i = lane; i < D; i += 32) orow[i] = (xr[i] - mu) * inv * g[i] + bb[i];
}

// mv[b][tau] = (1/D) * sum_t G[b][t][(t - tau) mod L]
__global__ void k_diagsum(const float* __restrict__ G, float* __restrict__ mv,
                          int L, float inv)
{
    int b = blockIdx.y, tau = blockIdx.x;
    const float* Gb = G + (size_t)b * L * L;
    float s = 0.f;
    for (int t = threadIdx.x; t < L; t += TPB) {
        int j = t - tau;
        if (j < 0) j += L;
        s += Gb[(size_t)t * L + j];
    }
    __shared__ float red[TPB];
    red[threadIdx.x] = s;
    __syncthreads();
    for (int o = TPB / 2; o > 0; o >>= 1) {
        if (threadIdx.x < o) red[threadIdx.x] += red[threadIdx.x + o];
        __syncthreads();
    }
    if (threadIdx.x == 0) mv[(size_t)b * L + tau] = red[0] * inv;
}

// per-batch top-6 (descending, lower index wins ties) + softmax
__global__ void k_topk(const float* __restrict__ mv, float* __restrict__ w,
                       int* __restrict__ dly, int L)
{
    int b = blockIdx.x;
    __shared__ float sv[TPB];
    __shared__ int   si[TPB];
    __shared__ int   chosen[6];
    __shared__ float cvals[6];
    const float* row = mv + (size_t)b * L;
    for (int kk = 0; kk < 6; kk++) {
        float best = -3.4e38f; int bi = 0x7fffffff;
        for (int l = threadIdx.x; l < L; l += TPB) {
            bool skip = false;
            for (int j = 0; j < kk; j++) if (chosen[j] == l) skip = true;
            if (skip) continue;
            float v = row[l];
            if (v > best || (v == best && l < bi)) { best = v; bi = l; }
        }
        sv[threadIdx.x] = best; si[threadIdx.x] = bi;
        __syncthreads();
        for (int s = TPB / 2; s > 0; s >>= 1) {
            if (threadIdx.x < s) {
                float ov = sv[threadIdx.x + s]; int oi = si[threadIdx.x + s];
                if (ov > sv[threadIdx.x] || (ov == sv[threadIdx.x] && oi < si[threadIdx.x])) {
                    sv[threadIdx.x] = ov; si[threadIdx.x] = oi;
                }
            }
            __syncthreads();
        }
        if (threadIdx.x == 0) { chosen[kk] = si[0]; cvals[kk] = sv[0]; }
        __syncthreads();
    }
    if (threadIdx.x == 0) {
        float m = cvals[0];
        for (int j = 1; j < 6; j++) m = fmaxf(m, cvals[j]);
        float e[6], s = 0.f;
        for (int j = 0; j < 6; j++) { e[j] = expf(cvals[j] - m); s += e[j]; }
        for (int j = 0; j < 6; j++) { w[b * 6 + j] = e[j] / s; dly[b * 6 + j] = chosen[j]; }
    }
}

// out[b][l][d] = sum_k w[b][k] * V[b][(l + delay[b][k]) mod Lmod][d]
__global__ void k_agg(const float* __restrict__ V, long long bsV,
                      const float* __restrict__ W, const int* __restrict__ Dly,
                      float* __restrict__ o, int Bn, int Lq, int Lmod, int D)
{
    long long idx = (long long)blockIdx.x * TPB + threadIdx.x;
    long long total = (long long)Bn * Lq * D;
    if (idx >= total) return;
    int d = (int)(idx % D);
    int l = (int)((idx / D) % Lq);
    int b = (int)(idx / ((long long)D * Lq));
    float s = 0.f;
    #pragma unroll
    for (int k = 0; k < 6; k++) {
        int t = l + Dly[b * 6 + k];
        if (t >= Lmod) t -= Lmod;
        s += W[b * 6 + k] * V[(size_t)b * bsV + (size_t)t * D + d];
    }
    o[idx] = s;
}

__global__ void k_add3(const float* a, const float* b, const float* c,
                       float* o, long long total)
{
    long long idx = (long long)blockIdx.x * TPB + threadIdx.x;
    if (idx >= total) return;
    o[idx] = a[idx] + b[idx] + c[idx];
}

// ---------------------------------------------------------------------------
static inline unsigned cdiv_ll(long long a, long long b) { return (unsigned)((a + b - 1) / b); }

static inline void gemm_nt(hipStream_t st,
                           const float* A, long long bsA,
                           const float* Bm, long long bsB,
                           const float* bias, const float* res, long long bsR,
                           float* C, long long bsC,
                           int M, int N, int K, int batch, int act)
{
    dim3 g((N + 127) / 128, (M + 127) / 128, batch);
    k_gemm_nt<<<g, dim3(TPB), 0, st>>>(A, bsA, Bm, bsB, bias, res, bsR, C, bsC, M, N, K, act);
}

extern "C" void kernel_launch(void* const* d_in, const int* in_sizes, int n_in,
                              void* d_out, int out_size, void* d_ws, size_t ws_size,
                              hipStream_t stream)
{
    (void)in_sizes; (void)n_in; (void)out_size; (void)ws_size;
    const int B = 16, L = 1024, NC = 128, D = 512, F = 2048, LD2 = 513, LAB = 512;

    const float* x_enc      = (const float*)d_in[0];
    const float* emb_enc_w  = (const float*)d_in[1];
    const float* emb_dec_w  = (const float*)d_in[2];
    const float* enc_attn_w = (const float*)d_in[3];
    const float* enc_attn_b = (const float*)d_in[4];
    const float* enc_ff1_w  = (const float*)d_in[5];
    const float* enc_ff2_w  = (const float*)d_in[6];
    const float* enc_norm_g = (const float*)d_in[7];
    const float* enc_norm_b = (const float*)d_in[8];
    const float* dec_self_w = (const float*)d_in[9];
    const float* dec_self_b = (const float*)d_in[10];
    const float* dec_cross_w= (const float*)d_in[11];
    const float* dec_cross_b= (const float*)d_in[12];
    const float* dec_ff1_w  = (const float*)d_in[13];
    const float* dec_ff2_w  = (const float*)d_in[14];
    const float* dec_trend_w= (const float*)d_in[15];
    const float* dec_norm_g = (const float*)d_in[16];
    const float* dec_norm_b = (const float*)d_in[17];
    const float* proj_w     = (const float*)d_in[18];
    const float* proj_b     = (const float*)d_in[19];
    float* outp = (float*)d_out;

    // --- workspace bump allocator (256B aligned) ---
    size_t off = 0;
    auto alloc_f = [&](size_t n) -> float* {
        float* p = (float*)((char*)d_ws + off);
        off += ((n * sizeof(float) + 255) / 256) * 256;
        return p;
    };
    float* t_init   = alloc_f((size_t)B * L * NC);
    float* s_init   = alloc_f((size_t)B * L * NC);
    float* xmean    = alloc_f((size_t)B * NC);
    float* trend    = alloc_f((size_t)B * LD2 * NC);
    float* seasonal = alloc_f((size_t)B * LD2 * NC);
    float* xs       = alloc_f((size_t)B * LD2 * 3 * D);   // covers all im2col shapes
    float* enc      = alloc_f((size_t)B * L * D);
    float* qb       = alloc_f((size_t)B * L * D);
    float* kb       = alloc_f((size_t)B * L * D);
    float* vb       = alloc_f((size_t)B * L * D);
    float* ac       = alloc_f((size_t)B * L * D);
    float* hb       = alloc_f((size_t)B * L * D);
    float* tmp      = alloc_f((size_t)B * L * D);
    float* ffb      = alloc_f((size_t)B * L * F);
    float* gram     = alloc_f((size_t)B * L * L);
    float* mv       = alloc_f((size_t)B * L);
    float* wts      = alloc_f((size_t)B * 6);
    int*   dly      = (int*)alloc_f((size_t)B * 6);
    float* colmu    = alloc_f((size_t)B * D);
    float* t1       = alloc_f((size_t)B * L * D);
    float* t2       = alloc_f((size_t)B * LD2 * D);
    float* t3       = alloc_f((size_t)B * LD2 * D);
    float* dx       = alloc_f((size_t)B * LD2 * D);

    const float invD = 1.0f / (float)D;
    long long tot;

    // ---- input decomposition & trend/seasonal init ----
    tot = (long long)B * L * NC;
    k_decomp<<<cdiv_ll(tot, TPB), TPB, 0, stream>>>(x_enc, s_init, t_init, B, L, NC);
    k_colmean<<<cdiv_ll(B * NC, TPB), TPB, 0, stream>>>(x_enc, xmean, B, L, NC);
    tot = (long long)B * LD2 * NC;
    k_init_td<<<cdiv_ll(tot, TPB), TPB, 0, stream>>>(t_init, s_init, xmean, trend, seasonal,
                                                     B, L, NC, LD2, LAB);

    // ---- encoder embedding: circ_conv3(x_enc, emb_enc_w) ----
    tot = (long long)B * L * NC * 3;
    k_build_xs<<<cdiv_ll(tot, TPB), TPB, 0, stream>>>(x_enc, xs, B, L, NC);
    gemm_nt(stream, xs, 0, emb_enc_w, 0, nullptr, nullptr, 0, enc, 0,
            B * L, D, NC * 3, 1, 0);

    // ---- encoder layers ----
    for (int l = 0; l < 2; l++) {
        const float* W = enc_attn_w + (size_t)l * 4 * D * D;
        const float* bs= enc_attn_b + (size_t)l * 4 * D;
        gemm_nt(stream, enc, 0, W + 0 * (size_t)D * D, 0, bs + 0 * D, nullptr, 0, qb, 0, B * L, D, D, 1, 0);
        gemm_nt(stream, enc, 0, W + 1 * (size_t)D * D, 0, bs + 1 * D, nullptr, 0, kb, 0, B * L, D, D, 1, 0);
        gemm_nt(stream, enc, 0, W + 2 * (size_t)D * D, 0, bs + 2 * D, nullptr, 0, vb, 0, B * L, D, D, 1, 0);
        // autocorrelation via per-batch Gram matrix
        gemm_nt(stream, qb, (long long)L * D, kb, (long long)L * D, nullptr, nullptr, 0,
                gram, (long long)L * L, L, L, D, B, 0);
        k_diagsum<<<dim3(L, B), TPB, 0, stream>>>(gram, mv, L, invD);
        k_topk<<<B, TPB, 0, stream>>>(mv, wts, dly, L);
        tot = (long long)B * L * D;
        k_agg<<<cdiv_ll(tot, TPB), TPB, 0, stream>>>(vb, (long long)L * D, wts, dly, ac, B, L, L, D);
        // out-proj + residual
        gemm_nt(stream, ac, 0, W + 3 * (size_t)D * D, 0, bs + 3 * D, enc, 0, hb, 0, B * L, D, D, 1, 0);
        k_decomp<<<cdiv_ll(tot, TPB), TPB, 0, stream>>>(hb, tmp, t1, B, L, D);  // tmp = h
        // FF
        gemm_nt(stream, tmp, 0, enc_ff1_w + (size_t)l * F * D, 0, nullptr, nullptr, 0,
                ffb, 0, B * L, F, D, 1, 1 /*gelu*/);
        gemm_nt(stream, ffb, 0, enc_ff2_w + (size_t)l * D * F, 0, nullptr, tmp, 0,
                hb, 0, B * L, D, F, 1, 0);
        k_decomp<<<cdiv_ll(tot, TPB), TPB, 0, stream>>>(hb, enc, t1, B, L, D);  // enc = seasonal
    }
    // my_ln(enc)
    k_ln<<<cdiv_ll((long long)B * L, 8), TPB, 0, stream>>>(enc, enc_norm_g, enc_norm_b, hb, B * L, D);
    k_colmean<<<cdiv_ll(B * D, TPB), TPB, 0, stream>>>(hb, colmu, B, L, D);
    tot = (long long)B * L * D;
    k_submean<<<cdiv_ll(tot, TPB), TPB, 0, stream>>>(hb, colmu, enc, B, L, D);

    // ---- decoder embedding ----
    tot = (long long)B * LD2 * NC * 3;
    k_build_xs<<<cdiv_ll(tot, TPB), TPB, 0, stream>>>(seasonal, xs, B, LD2, NC);
    gemm_nt(stream, xs, 0, emb_dec_w, 0, nullptr, nullptr, 0, dx, 0, B * LD2, D, NC * 3, 1, 0);

    const long long totd = (long long)B * LD2 * D;

    // ---- decoder layer 0: self attention ----
    gemm_nt(stream, dx, 0, dec_self_w + 0 * (size_t)D * D, 0, dec_self_b + 0 * D, nullptr, 0, qb, 0, B * LD2, D, D, 1, 0);
    gemm_nt(stream, dx, 0, dec_self_w + 1 * (size_t)D * D, 0, dec_self_b + 1 * D, nullptr, 0, kb, 0, B * LD2, D, D, 1, 0);
    gemm_nt(stream, dx, 0, dec_self_w + 2 * (size_t)D * D, 0, dec_self_b + 2 * D, nullptr, 0, vb, 0, B * LD2, D, D, 1, 0);
    gemm_nt(stream, qb, (long long)LD2 * D, kb, (long long)LD2 * D, nullptr, nullptr, 0,
            gram, (long long)LD2 * LD2, LD2, LD2, D, B, 0);
    k_diagsum<<<dim3(LD2, B), TPB, 0, stream>>>(gram, mv, LD2, invD);
    k_topk<<<B, TPB, 0, stream>>>(mv, wts, dly, LD2);
    k_agg<<<cdiv_ll(totd, TPB), TPB, 0, stream>>>(vb, (long long)LD2 * D, wts, dly, ac, B, LD2, LD2, D);
    gemm_nt(stream, ac, 0, dec_self_w + 3 * (size_t)D * D, 0, dec_self_b + 3 * D, dx, 0, hb, 0, B * LD2, D, D, 1, 0);
    k_decomp<<<cdiv_ll(totd, TPB), TPB, 0, stream>>>(hb, dx, t1, B, LD2, D);

    // ---- cross attention (k,v from encoder output, truncated to LD2 rows/batch) ----
    gemm_nt(stream, dx, 0, dec_cross_w + 0 * (size_t)D * D, 0, dec_cross_b + 0 * D, nullptr, 0, qb, 0, B * LD2, D, D, 1, 0);
    gemm_nt(stream, enc, 0, dec_cross_w + 1 * (size_t)D * D, 0, dec_cross_b + 1 * D, nullptr, 0, kb, 0, B * L, D, D, 1, 0);
    gemm_nt(stream, enc, 0, dec_cross_w + 2 * (size_t)D * D, 0, dec_cross_b + 2 * D, nullptr, 0, vb, 0, B * L, D, D, 1, 0);
    gemm_nt(stream, qb, (long long)LD2 * D, kb, (long long)L * D, nullptr, nullptr, 0,
            gram, (long long)LD2 * LD2, LD2, LD2, D, B, 0);
    k_diagsum<<<dim3(LD2, B), TPB, 0, stream>>>(gram, mv, LD2, invD);
    k_topk<<<B, TPB, 0, stream>>>(mv, wts, dly, LD2);
    k_agg<<<cdiv_ll(totd, TPB), TPB, 0, stream>>>(vb, (long long)L * D, wts, dly, ac, B, LD2, LD2, D);
    gemm_nt(stream, ac, 0, dec_cross_w + 3 * (size_t)D * D, 0, dec_cross_b + 3 * D, dx, 0, hb, 0, B * LD2, D, D, 1, 0);
    k_decomp<<<cdiv_ll(totd, TPB), TPB, 0, stream>>>(hb, dx, t2, B, LD2, D);

    // ---- decoder FF ----
    gemm_nt(stream, dx, 0, dec_ff1_w, 0, nullptr, nullptr, 0, ffb, 0, B * LD2, F, D, 1, 1);
    gemm_nt(stream, ffb, 0, dec_ff2_w, 0, nullptr, dx, 0, hb, 0, B * LD2, D, F, 1, 0);
    k_decomp<<<cdiv_ll(totd, TPB), TPB, 0, stream>>>(hb, dx, t3, B, LD2, D);

    // ---- trend accumulation: trend += circ_conv3(t1+t2+t3, dec_trend_w) ----
    k_add3<<<cdiv_ll(totd, TPB), TPB, 0, stream>>>(t1, t2, t3, tmp, totd);
    tot = (long long)B * LD2 * D * 3;
    k_build_xs<<<cdiv_ll(tot, TPB), TPB, 0, stream>>>(tmp, xs, B, LD2, D);
    gemm_nt(stream, xs, 0, dec_trend_w, 0, nullptr, trend, 0, trend, 0, B * LD2, NC, 3 * D, 1, 0);

    // ---- final my_ln(dx) ----
    k_ln<<<cdiv_ll((long long)B * LD2, 8), TPB, 0, stream>>>(dx, dec_norm_g, dec_norm_b, hb, B * LD2, D);
    k_colmean<<<cdiv_ll(B * D, TPB), TPB, 0, stream>>>(hb, colmu, B, LD2, D);
    k_submean<<<cdiv_ll(totd, TPB), TPB, 0, stream>>>(hb, colmu, tmp, B, LD2, D);

    // ---- output: last row per batch: out = trend[-1] + dx[-1] @ proj_w^T + proj_b ----
    gemm_nt(stream,
            tmp + (size_t)(LD2 - 1) * D, (long long)LD2 * D,
            proj_w, 0,
            proj_b,
            trend + (size_t)(LD2 - 1) * NC, (long long)LD2 * NC,
            outp, (long long)NC,
            1, NC, D, B, 0);
}
